// QRNNModel_22256520528464
// MI455X (gfx1250) — compile-verified, matrix operand order checked by reading, not run on hardware
//
#include <hip/hip_runtime.h>
#include <hip/hip_bf16.h>

typedef unsigned short u16;
typedef unsigned int   u32;
typedef unsigned long long u64;

typedef __attribute__((ext_vector_type(16))) __bf16 v16bf;
typedef __attribute__((ext_vector_type(8)))  float  v8f;
typedef __attribute__((ext_vector_type(4)))  unsigned int uv4;
typedef __attribute__((ext_vector_type(8)))  int iv8;
typedef __attribute__((ext_vector_type(4)))  int iv4;

// B=32, S=1024, IN=1024, D=1024, MLP=2048, NC=10
#define ROWS   32768          // B*S
#define KDIM   1024           // IN
#define NDIM   2048           // D (f cols) + D (v cols)
#define DDIM   1024
#define MLPD   2048
#define NCLS   10

__device__ __forceinline__ u16 f2bf(float f) {
    u32 u = __float_as_uint(f);
    u32 r = u + 0x7FFFu + ((u >> 16) & 1u);   // RNE
    return (u16)(r >> 16);
}
__device__ __forceinline__ float sigm(float x) {
    return 1.0f / (1.0f + __expf(-x));
}

// ---------------- 1) convert x (f32) -> bf16 --------------------------------
__global__ void cvt_x_kernel(const float4* __restrict__ x, u32* __restrict__ xb2, int n4) {
    int i = blockIdx.x * blockDim.x + threadIdx.x;
    if (i >= n4) return;
    float4 v = x[i];
    u32 lo = (u32)f2bf(v.x) | ((u32)f2bf(v.y) << 16);
    u32 hi = (u32)f2bf(v.z) | ((u32)f2bf(v.w) << 16);
    xb2[2 * i]     = lo;
    xb2[2 * i + 1] = hi;
}

// -------- 2) pack WcT[n][k] = transpose of [ W_w[:,0:1024] | V_w[:,0:1024] ]
__global__ void build_wcat_kernel(const float* __restrict__ W,
                                  const float* __restrict__ V,
                                  u16* __restrict__ WcT) {      // [2048][1024] bf16
    int idx = blockIdx.x * blockDim.x + threadIdx.x;            // 2048*1024
    if (idx >= KDIM * NDIM) return;
    int n = idx >> 10;          // 0..2047 output row
    int k = idx & 1023;         // 0..1023
    float w = (n < DDIM) ? W[(size_t)k * 3072 + n]
                         : V[(size_t)k * 3072 + (n - DDIM)];
    WcT[idx] = f2bf(w);
}

// ---------------- TDM: load a [128 rows x 32 bf16] tile into LDS ------------
// 2-D tensor, data_size=2B, dim0 len/stride = 1024 elems, tile 32x128,
// pad 4 DWORDs every 16 DWORDs -> LDS row stride = 80B (16B aligned).
__device__ __forceinline__ void tdm_load_tile(u32 lds_byte_addr, u64 gaddr, u32 dim1) {
    uv4 g0;
    g0.x = 1u;                                   // count=1, is_restore=0, gather off
    g0.y = lds_byte_addr;                        // lds_addr
    g0.z = (u32)gaddr;                           // global_addr[31:0]
    g0.w = (u32)(gaddr >> 32) | (2u << 30);      // global_addr[56:32] | type=2
    iv8 g1;
    g1[0] = (int)((1u << 16)      // data_size = 2 bytes
                | (1u << 20)      // pad_enable
                | (3u << 22)      // pad_interval code 3 -> every 16 DWORDs
                | (3u << 25));    // pad_amount   code 3 -> 4 DWORDs (16B)
    g1[1] = (int)(1024u << 16);   // tensor_dim0[15:0] = 1024 (at bits 63:48)
    g1[2] = (int)(dim1 << 16);    // tensor_dim1[15:0] (at bits 95:80)
    g1[3] = (int)(32u << 16);     // tile_dim0 = 32 (at bits 127:112)
    g1[4] = 128;                  // tile_dim1 = 128 (bits 143:128)
    g1[5] = 1024;                 // tensor_dim0_stride[31:0] (bits 191:160)
    g1[6] = 0;
    g1[7] = 0;
    iv4 z4 = (iv4)0;
#if __clang_major__ >= 23
    __builtin_amdgcn_tensor_load_to_lds(g0, g1, z4, z4, (iv8)0, 0);
#else
    __builtin_amdgcn_tensor_load_to_lds(g0, g1, z4, z4, 0);
#endif
}

// ---------------- 3) bf16 WMMA GEMM: pre[32768][2048] = xb @ WcT^T ----------
// block tile 128x128, BK=32, 256 threads (8 waves), wave tile 64x32 (4x2 WMMA)
// TDM double-buffered tile staging; one barrier per K-step.
union AFrag { u32 u[8]; uint4 q[2]; v16bf v; };

__global__ __launch_bounds__(256) void gemm_bf16_kernel(const u16* __restrict__ A,    // [32768][1024] bf16
                                                        const u16* __restrict__ Bt,   // [2048][1024]  bf16
                                                        float* __restrict__ C) {      // [32768][2048] f32
    __shared__ uint4 AsV[2][128 * 5];   // 128 rows x (16 data + 4 pad) DWORDs
    __shared__ uint4 BsV[2][128 * 5];

    const int tid   = threadIdx.x;
    const int lane  = tid & 31;
    const int wave  = tid >> 5;        // 0..7
    const int waveM = wave >> 2;       // 0..1  (64 rows each)
    const int waveN = wave & 3;        // 0..3  (32 cols each)
    const int half  = lane >> 4;       // K-half selector
    const int row16 = lane & 15;

    const int m0 = blockIdx.y * 128;
    const int n0 = blockIdx.x * 128;

    const u64 Abase = (u64)A  + (u64)m0 * KDIM * 2ull;
    const u64 Bbase = (u64)Bt + (u64)n0 * KDIM * 2ull;

    v8f acc[4][2];
#pragma unroll
    for (int i = 0; i < 4; ++i)
#pragma unroll
        for (int j = 0; j < 2; ++j) acc[i][j] = (v8f)0.0f;

    if (wave == 0) {
        tdm_load_tile((u32)(u64)&AsV[0][0], Abase, 32768u);
        tdm_load_tile((u32)(u64)&BsV[0][0], Bbase, 2048u);
    }

    int cur = 0;
    for (int k0 = 0; k0 < KDIM; k0 += 32) {
        if (wave == 0) __builtin_amdgcn_s_wait_tensorcnt(0);  // tile k0 resident
        __syncthreads();                                      // all waves see it;
                                                              // prev reads of other buf done
        if (wave == 0 && (k0 + 32) < KDIM) {                  // prefetch next tile
            u64 off = (u64)(k0 + 32) * 2ull;
            tdm_load_tile((u32)(u64)&AsV[cur ^ 1][0], Abase + off, 32768u);
            tdm_load_tile((u32)(u64)&BsV[cur ^ 1][0], Bbase + off, 2048u);
        }

        AFrag af[4], bfr[2];
#pragma unroll
        for (int i = 0; i < 4; ++i) {
            const uint4* Ap = &AsV[cur][(waveM * 64 + i * 16 + row16) * 5];
            af[i].q[0] = Ap[half];          // K 0..15 half (4 DWORDs, 16B aligned)
            af[i].q[1] = Ap[2 + half];      // K 16..31 half
        }
#pragma unroll
        for (int j = 0; j < 2; ++j) {
            const uint4* Bp = &BsV[cur][(waveN * 32 + j * 16 + row16) * 5];
            bfr[j].q[0] = Bp[half];
            bfr[j].q[1] = Bp[2 + half];
        }
#pragma unroll
        for (int i = 0; i < 4; ++i)
#pragma unroll
            for (int j = 0; j < 2; ++j)
                acc[i][j] = __builtin_amdgcn_wmma_f32_16x16x32_bf16(
                    false, af[i].v, false, bfr[j].v, (short)0, acc[i][j], false, false);
        cur ^= 1;
    }

    // epilogue: C/D layout -> VGPR q holds row (q + 8*half), col = lane&15
#pragma unroll
    for (int i = 0; i < 4; ++i) {
        int rbase = m0 + waveM * 64 + i * 16;
#pragma unroll
        for (int j = 0; j < 2; ++j) {
            int c = n0 + waveN * 32 + j * 16 + row16;
#pragma unroll
            for (int q = 0; q < 8; ++q) {
                int r = rbase + q + 8 * half;
                C[(size_t)r * NDIM + c] = acc[i][j][q];
            }
        }
    }
}

// ---------------- 4) o at last timestep only --------------------------------
__global__ void o_last_kernel(const float* __restrict__ x,   // [32][1024][1024]
                              const float* __restrict__ W,   // [1024][3072]
                              const float* __restrict__ V,   // [1024][3072]
                              const float* __restrict__ Vb,  // [3072]
                              float* __restrict__ o) {       // [32][1024]
    int t = blockIdx.x * blockDim.x + threadIdx.x;           // 32768
    int b = t >> 10, d = t & 1023;
    const float* x1 = x + ((size_t)b * 1024 + 1023) * 1024;  // x[b, S-1]
    const float* x0 = x + ((size_t)b * 1024 + 1022) * 1024;  // x[b, S-2]
    float s = Vb[2048 + d];
    for (int i = 0; i < KDIM; ++i) {
        s += x1[i] * W[(size_t)i * 3072 + 2048 + d];
        s += x0[i] * V[(size_t)i * 3072 + 2048 + d];
    }
    o[t] = sigm(s);
}

// ---------------- 5) sequential scan over S + multiply by o ------------------
__global__ void scan_kernel(const float* __restrict__ pre,  // [32768][2048]
                            const float* __restrict__ Vb,
                            const float* __restrict__ o,    // [32][1024]
                            float* __restrict__ h) {        // [32][1024]
    int t = blockIdx.x * blockDim.x + threadIdx.x;          // 32768
    int b = t >> 10, d = t & 1023;
    const float* base = pre + (size_t)b * 1024 * NDIM;
    float bias = Vb[d];
    float c = 0.0f, vprev = 0.0f;
    for (int s = 0; s < 1024; ++s) {
        float u = base[(size_t)s * NDIM + d];
        float f = sigm(u + vprev + bias);
        c = c * f + f;
        vprev = base[(size_t)s * NDIM + 1024 + d];          // v[b,s] feeds step s+1
    }
    h[t] = c * o[t];
}

// ---------------- 6-8) tiny MLP head ----------------------------------------
__global__ void mlp0_kernel(const float* __restrict__ h, const float* __restrict__ w,
                            const float* __restrict__ b, float* __restrict__ q) {
    int t = blockIdx.x * blockDim.x + threadIdx.x;          // 32*2048
    int bi = t >> 11, n = t & 2047;
    const float* hr = h + bi * DDIM;
    float s = b[n];
    for (int i = 0; i < DDIM; ++i) s += hr[i] * w[(size_t)i * MLPD + n];
    q[t] = fmaxf(s, 0.0f);
}
__global__ void mlp1_kernel(const float* __restrict__ q1, const float* __restrict__ w,
                            const float* __restrict__ b, float* __restrict__ q) {
    int t = blockIdx.x * blockDim.x + threadIdx.x;          // 32*2048
    int bi = t >> 11, n = t & 2047;
    const float* qr = q1 + bi * MLPD;
    float s = b[n];
    for (int i = 0; i < MLPD; ++i) s += qr[i] * w[(size_t)i * MLPD + n];
    q[t] = fmaxf(s, 0.0f);
}
__global__ void mlp2_kernel(const float* __restrict__ q2, const float* __restrict__ w,
                            const float* __restrict__ b, float* __restrict__ out) {
    int t = blockIdx.x * blockDim.x + threadIdx.x;          // 320
    if (t >= 32 * NCLS) return;
    int bi = t / NCLS, n = t % NCLS;
    const float* qr = q2 + bi * MLPD;
    float s = b[n];
    for (int i = 0; i < MLPD; ++i) s += qr[i] * w[(size_t)i * NCLS + n];
    out[t] = s;
}

// ---------------- launcher ---------------------------------------------------
extern "C" void kernel_launch(void* const* d_in, const int* in_sizes, int n_in,
                              void* d_out, int out_size, void* d_ws, size_t ws_size,
                              hipStream_t stream) {
    const float* x    = (const float*)d_in[0];
    const float* W_w  = (const float*)d_in[1];
    const float* V_w  = (const float*)d_in[2];
    const float* V_b  = (const float*)d_in[3];
    const float* l0_w = (const float*)d_in[4];
    const float* l0_b = (const float*)d_in[5];
    const float* l1_w = (const float*)d_in[6];
    const float* l1_b = (const float*)d_in[7];
    const float* l2_w = (const float*)d_in[8];
    const float* l2_b = (const float*)d_in[9];
    float* out = (float*)d_out;

    char* ws = (char*)d_ws;
    u16*   xb  = (u16*)ws;                                   // 64 MB
    u16*   WcT = (u16*)(ws + (size_t)64 * 1024 * 1024);      //  4 MB
    float* pre = (float*)(ws + (size_t)68 * 1024 * 1024);    // 256 MB
    float* oB  = (float*)(ws + (size_t)324 * 1024 * 1024);   // 128 KB
    float* h   = oB + 32768;
    float* q1  = h + 32768;
    float* q2  = q1 + 32 * MLPD;

    // 1) x -> bf16
    {
        int n4 = (ROWS * KDIM) / 4;                          // 8388608
        cvt_x_kernel<<<n4 / 256, 256, 0, stream>>>((const float4*)x, (u32*)xb, n4);
    }
    // 2) pack transposed weights
    build_wcat_kernel<<<(KDIM * NDIM) / 256, 256, 0, stream>>>(W_w, V_w, WcT);
    // 3) big bf16 WMMA GEMM with TDM-staged tiles
    {
        dim3 grid(NDIM / 128, ROWS / 128);                   // (16, 256)
        gemm_bf16_kernel<<<grid, 256, 0, stream>>>(xb, WcT, pre);
    }
    // 4) o at final timestep
    o_last_kernel<<<32768 / 256, 256, 0, stream>>>(x, W_w, V_w, V_b, oB);
    // 5) recurrence scan + h_last
    scan_kernel<<<32768 / 256, 256, 0, stream>>>(pre, V_b, oB, h);
    // 6-8) MLP head
    mlp0_kernel<<<(32 * MLPD) / 256, 256, 0, stream>>>(h, l0_w, l0_b, q1);
    mlp1_kernel<<<(32 * MLPD) / 256, 256, 0, stream>>>(q1, l1_w, l1_b, q2);
    mlp2_kernel<<<2, 256, 0, stream>>>(q2, l2_w, l2_b, out);
    (void)in_sizes; (void)n_in; (void)out_size; (void)ws_size;
}
